// GraphonNewEncoder_22110491639897
// MI455X (gfx1250) — compile-verified, matrix operand order checked by reading, not run on hardware
//
#include <hip/hip_runtime.h>

// ---------------- problem constants (match reference) ----------------
#define NN   100000
#define FF   128
#define HH   256
#define OUTD 128
#define GG   512
#define EE   1600000
#define BN_EPS 1e-5f
#define ROWCHUNK 512

// ---------------- WMMA types ----------------
typedef __attribute__((ext_vector_type(16))) __bf16        v16bf;
typedef __attribute__((ext_vector_type(8)))  float         v8f;
typedef __attribute__((ext_vector_type(4)))  unsigned int  uint4v;
typedef __attribute__((ext_vector_type(4)))  unsigned short ushort4v;

union FragBF { v16bf v; uint4v u[2]; };

__device__ __forceinline__ unsigned short f2bf(float f) {
    unsigned int u = __builtin_bit_cast(unsigned int, f);
    unsigned int r = u + 0x7fffu + ((u >> 16) & 1u);   // round-to-nearest-even
    return (unsigned short)(r >> 16);
}

// ---------------- 1) agg = x (vectorized copy) ----------------
__global__ void __launch_bounds__(256) copy_f4(const float* __restrict__ x,
                                               float* __restrict__ agg, int n4) {
    int i = blockIdx.x * 256 + threadIdx.x;
    if (i < n4) ((float4*)agg)[i] = ((const float4*)x)[i];
}

// ---------------- 2) agg[dst] += x[src] over edges ----------------
__global__ void __launch_bounds__(256) scatter_edges(const float* __restrict__ x,
                                                     const int* __restrict__ src,
                                                     const int* __restrict__ dst,
                                                     float* __restrict__ agg) {
    unsigned long long tid = (unsigned long long)blockIdx.x * 256ull + threadIdx.x;
    int e = (int)(tid >> 5);
    if (e >= EE) return;
    int f4 = ((int)tid & 31) * 4;                 // 32 lanes * 4 floats = 128 feats
    int s = src[e], d = dst[e];
    float4 v = *(const float4*)(x + (size_t)s * FF + f4);
    float* o = agg + (size_t)d * FF + f4;
    atomicAdd(o + 0, v.x); atomicAdd(o + 1, v.y);
    atomicAdd(o + 2, v.z); atomicAdd(o + 3, v.w);
}

// ---------------- 3a) agg f32 -> bf16 ----------------
__global__ void __launch_bounds__(256) convert_bf16(const float* __restrict__ in,
                                                    unsigned short* __restrict__ out, int n4) {
    int i = blockIdx.x * 256 + threadIdx.x;
    if (i >= n4) return;
    float4 v = ((const float4*)in)[i];
    ushort4v o;
    o.x = f2bf(v.x); o.y = f2bf(v.y); o.z = f2bf(v.z); o.w = f2bf(v.w);
    ((ushort4v*)out)[i] = o;
}

// ---------------- 3b) w_mlp [F,H] -> wT bf16 [H,F] (column-major B) ----------------
__global__ void __launch_bounds__(256) w_transpose_bf16(const float* __restrict__ w,
                                                        unsigned short* __restrict__ wT) {
    int idx = blockIdx.x * 256 + threadIdx.x;       // F*H = 32768
    int k = idx >> 8;                               // 0..127
    int n = idx & 255;                              // 0..255 (coalesced read)
    wT[(size_t)n * FF + k] = f2bf(w[idx]);
}

// ---------------- 4) WMMA GEMM: hlin[N,256] = agg_bf[N,128] @ w_bf + b ----------------
// Block = 256 threads = 8 waves; block covers 16 rows x 256 cols; wave w -> cols [32w,32w+32)
__global__ void __launch_bounds__(256) gemm_bf16(const unsigned short* __restrict__ A,
                                                 const unsigned short* __restrict__ BT,
                                                 const float* __restrict__ bias,
                                                 float* __restrict__ C) {
    const int lane = threadIdx.x & 31;
    const int wave = threadIdx.x >> 5;
    const int r    = lane & 15;
    const int hi   = lane >> 4;

    const int row  = blockIdx.x * 16 + r;           // A-frag row for this lane
    const int col0 = wave * 32 + r;                 // B-frag col (tile 0)
    const unsigned short* arow = A  + (size_t)row  * FF;
    const unsigned short* bp0  = BT + (size_t)col0 * FF;
    const unsigned short* bp1  = bp0 + 16 * FF;

    v8f c0 = {}; v8f c1 = {};
#pragma unroll
    for (int kb = 0; kb < FF; kb += 32) {
        FragBF a, b0, b1;
        // 16-bit A layout: lanes 0-15 hold K=[0..7],[16..23]; lanes 16-31 hold K=[8..15],[24..31]
        a.u[0]  = *(const uint4v*)(arow + kb +      hi * 8);
        a.u[1]  = *(const uint4v*)(arow + kb + 16 + hi * 8);
        b0.u[0] = *(const uint4v*)(bp0  + kb +      hi * 8);
        b0.u[1] = *(const uint4v*)(bp0  + kb + 16 + hi * 8);
        b1.u[0] = *(const uint4v*)(bp1  + kb +      hi * 8);
        b1.u[1] = *(const uint4v*)(bp1  + kb + 16 + hi * 8);
        c0 = __builtin_amdgcn_wmma_f32_16x16x32_bf16(false, a.v, false, b0.v,
                                                     (short)0, c0, false, false);
        c1 = __builtin_amdgcn_wmma_f32_16x16x32_bf16(false, a.v, false, b1.v,
                                                     (short)0, c1, false, false);
    }
    // C/D layout: VGPR j -> row = 16*tile + j + 8*hi, col = 16*tile_n + (lane&15)
    const int baseRow = blockIdx.x * 16 + hi * 8;
    const float bia0 = bias[col0], bia1 = bias[col0 + 16];
#pragma unroll
    for (int j = 0; j < 8; ++j) {
        size_t ro = (size_t)(baseRow + j) * HH;
        C[ro + col0]      = c0[j] + bia0;
        C[ro + col0 + 16] = c1[j] + bia1;
    }
}

// ---------------- 5) column sums / sums-of-squares over [N,256] ----------------
__global__ void __launch_bounds__(256) col_stats(const float* __restrict__ h,
                                                 float* __restrict__ sum,
                                                 float* __restrict__ sumsq) {
    int c  = threadIdx.x;
    int r0 = blockIdx.x * ROWCHUNK;
    int r1 = min(r0 + ROWCHUNK, NN);
    float s = 0.f, q = 0.f;
    for (int r = r0; r < r1; ++r) {
        float v = h[(size_t)r * HH + c];
        s += v; q += v * v;
    }
    atomicAdd(&sum[c], s); atomicAdd(&sumsq[c], q);
}

// ---------------- 6) finalize BN affine: y = x*scale + shift ----------------
__global__ void bn_finalize(const float* __restrict__ sum, const float* __restrict__ sumsq,
                            const float* __restrict__ g, const float* __restrict__ b,
                            float* __restrict__ scale, float* __restrict__ shift) {
    int c = threadIdx.x;
    const float inv_n = 1.0f / (float)NN;
    float mu  = sum[c] * inv_n;
    float var = sumsq[c] * inv_n - mu * mu;          // biased var, matches jnp.var
    float is  = rsqrtf(var + BN_EPS);
    float sc  = g[c] * is;
    scale[c] = sc;
    shift[c] = b[c] - mu * sc;
}

// ---------------- 7) BN1+ReLU in place, accumulate stats for BN2 ----------------
__global__ void __launch_bounds__(256) bn_relu_stats(float* __restrict__ h,
                                                     const float* __restrict__ scale,
                                                     const float* __restrict__ shift,
                                                     float* __restrict__ sum2,
                                                     float* __restrict__ sumsq2) {
    int c = threadIdx.x;
    float sc = scale[c], sh = shift[c];
    int r0 = blockIdx.x * ROWCHUNK;
    int r1 = min(r0 + ROWCHUNK, NN);
    float s = 0.f, q = 0.f;
    for (int r = r0; r < r1; ++r) {
        size_t idx = (size_t)r * HH + c;
        float v = fmaxf(h[idx] * sc + sh, 0.f);
        h[idx] = v;
        s += v; q += v * v;
    }
    atomicAdd(&sum2[c], s); atomicAdd(&sumsq2[c], q);
}

// ---------------- 8) BN2+ReLU, segment-sum into pooled1 (gids sorted -> run-length) ----
__global__ void __launch_bounds__(256) bn_relu_pool(const float* __restrict__ h,
                                                    const float* __restrict__ scale,
                                                    const float* __restrict__ shift,
                                                    const int* __restrict__ gids,
                                                    float* __restrict__ pooled1) {
    int c = threadIdx.x;
    float sc = scale[c], sh = shift[c];
    int r0 = blockIdx.x * ROWCHUNK;
    int r1 = min(r0 + ROWCHUNK, NN);
    int cur = -1; float acc = 0.f;
    for (int r = r0; r < r1; ++r) {
        float v = fmaxf(h[(size_t)r * HH + c] * sc + sh, 0.f);
        int g = gids[r];
        if (g != cur) {
            if (cur >= 0) atomicAdd(&pooled1[(size_t)cur * HH + c], acc);
            cur = g; acc = 0.f;
        }
        acc += v;
    }
    if (cur >= 0) atomicAdd(&pooled1[(size_t)cur * HH + c], acc);
}

// ---------------- 9) pooled0 = segment_sum(x) ----------------
__global__ void __launch_bounds__(128) pool_x(const float* __restrict__ x,
                                              const int* __restrict__ gids,
                                              float* __restrict__ pooled0) {
    int c = threadIdx.x;                            // 0..127
    int r0 = blockIdx.x * ROWCHUNK;
    int r1 = min(r0 + ROWCHUNK, NN);
    int cur = -1; float acc = 0.f;
    for (int r = r0; r < r1; ++r) {
        float v = x[(size_t)r * FF + c];
        int g = gids[r];
        if (g != cur) {
            if (cur >= 0) atomicAdd(&pooled0[(size_t)cur * FF + c], acc);
            cur = g; acc = 0.f;
        }
        acc += v;
    }
    if (cur >= 0) atomicAdd(&pooled0[(size_t)cur * FF + c], acc);
}

// ---------------- 10) score = pooled0@w0 + b0 + pooled1@w1 + b1 ----------------
__global__ void __launch_bounds__(128) score_kernel(const float* __restrict__ pooled0,
                                                    const float* __restrict__ pooled1,
                                                    const float* __restrict__ w0,
                                                    const float* __restrict__ b0,
                                                    const float* __restrict__ w1,
                                                    const float* __restrict__ b1,
                                                    float* __restrict__ out) {
    int o = threadIdx.x;                            // 0..127
    int g = blockIdx.x;                             // 0..511
    const float* p0 = pooled0 + (size_t)g * FF;
    const float* p1 = pooled1 + (size_t)g * HH;
    float s = b0[o] + b1[o];
    for (int k = 0; k < FF; ++k) s += p0[k] * w0[(size_t)k * OUTD + o];
    for (int k = 0; k < HH; ++k) s += p1[k] * w1[(size_t)k * OUTD + o];
    out[(size_t)g * OUTD + o] = s;
}

// ---------------- host launcher ----------------
extern "C" void kernel_launch(void* const* d_in, const int* in_sizes, int n_in,
                              void* d_out, int out_size, void* d_ws, size_t ws_size,
                              hipStream_t stream) {
    const float* x      = (const float*)d_in[0];
    const float* w_mlp  = (const float*)d_in[1];
    const float* b_mlp  = (const float*)d_in[2];
    const float* bn1_g  = (const float*)d_in[3];
    const float* bn1_b  = (const float*)d_in[4];
    const float* bn2_g  = (const float*)d_in[5];
    const float* bn2_b  = (const float*)d_in[6];
    const float* w_lin0 = (const float*)d_in[7];
    const float* b_lin0 = (const float*)d_in[8];
    const float* w_lin1 = (const float*)d_in[9];
    const float* b_lin1 = (const float*)d_in[10];
    const int*   eidx   = (const int*)d_in[11];
    const int*   gids   = (const int*)d_in[12];
    const int* src = eidx;
    const int* dst = eidx + EE;
    float* out = (float*)d_out;

    // workspace layout (256B-aligned regions)
    char* ws = (char*)d_ws;
    float* sum1    = (float*)(ws + 0);
    float* sumsq1  = (float*)(ws + 1024);
    float* sum2    = (float*)(ws + 2048);
    float* sumsq2  = (float*)(ws + 3072);
    float* scale1  = (float*)(ws + 4096);
    float* shift1  = (float*)(ws + 5120);
    float* scale2  = (float*)(ws + 6144);
    float* shift2  = (float*)(ws + 7168);
    float* pooled0 = (float*)(ws + 8192);                       // G*F   = 262144 B
    float* pooled1 = (float*)(ws + 8192 + 262144);              // G*H   = 524288 B
    size_t zero_bytes = 8192 + 262144 + 524288;                 // 794624
    float* agg     = (float*)(ws + zero_bytes);                 // N*F f32 = 51.2 MB
    float* hlin    = (float*)(ws + zero_bytes + 51200000ull);   // N*H f32 = 102.4 MB
    unsigned short* agg_bf = (unsigned short*)(ws + zero_bytes + 153600000ull); // 25.6 MB
    unsigned short* wT_bf  = (unsigned short*)(ws + zero_bytes + 179200000ull); // 64 KB

    // zero accumulators (graph-capture safe)
    hipMemsetAsync(d_ws, 0, zero_bytes, stream);

    // agg = x
    copy_f4<<<(NN * FF / 4 + 255) / 256, 256, 0, stream>>>(x, agg, NN * FF / 4);
    // agg[dst] += x[src]
    {
        unsigned long long thr = (unsigned long long)EE * 32ull;
        scatter_edges<<<(unsigned)((thr + 255) / 256), 256, 0, stream>>>(x, src, dst, agg);
    }
    // bf16 conversions
    convert_bf16<<<(NN * FF / 4 + 255) / 256, 256, 0, stream>>>(agg, agg_bf, NN * FF / 4);
    w_transpose_bf16<<<(FF * HH) / 256, 256, 0, stream>>>(w_mlp, wT_bf);
    // hlin = agg @ w_mlp + b_mlp   (WMMA bf16 -> f32)
    gemm_bf16<<<NN / 16, 256, 0, stream>>>(agg_bf, wT_bf, b_mlp, hlin);
    // BN1 stats + finalize
    int statBlocks = (NN + ROWCHUNK - 1) / ROWCHUNK;
    col_stats<<<statBlocks, 256, 0, stream>>>(hlin, sum1, sumsq1);
    bn_finalize<<<1, HH, 0, stream>>>(sum1, sumsq1, bn1_g, bn1_b, scale1, shift1);
    // BN1 + ReLU in place, gather BN2 stats
    bn_relu_stats<<<statBlocks, 256, 0, stream>>>(hlin, scale1, shift1, sum2, sumsq2);
    bn_finalize<<<1, HH, 0, stream>>>(sum2, sumsq2, bn2_g, bn2_b, scale2, shift2);
    // BN2 + ReLU fused with pooled1 segment sum
    bn_relu_pool<<<statBlocks, 256, 0, stream>>>(hlin, scale2, shift2, gids, pooled1);
    // pooled0 segment sum of x
    pool_x<<<statBlocks, 128, 0, stream>>>(x, gids, pooled0);
    // final heads
    score_kernel<<<GG, OUTD, 0, stream>>>(pooled0, pooled1, w_lin0, b_lin0,
                                          w_lin1, b_lin1, out);
}
